// compAggWikiqa_34694745817776
// MI455X (gfx1250) — matrix operand order repaired and checked by reading, not compiled, not run on hardware
//
#include <hip/hip_runtime.h>
#include <hip/hip_fp16.h>

// CompAgg-WikiQA on gfx1250 (MI455X). All GEMM-shaped work lowered to
// v_wmma_f32_16x16x32_f16 (wave32). Projection and conv stage their shared
// A-operand rows into LDS via GLOBAL_LOAD_ASYNC_TO_LDS_B128 (ASYNCcnt) so all
// 8 waves of a workgroup feed WMMA from ds_load_b128; B streams from L2.

typedef _Float16 half_t;
typedef __attribute__((ext_vector_type(16))) _Float16 v16h;
typedef __attribute__((ext_vector_type(4)))  _Float16 v4h;
typedef __attribute__((ext_vector_type(8)))  float    v8f;
typedef __attribute__((ext_vector_type(4)))  unsigned int v4u;
typedef __attribute__((ext_vector_type(4)))  int v4i;

union V16U { v16h h; v4u q[2]; };

#define T_TOK 32768
#define DIM   1024

#if __has_builtin(__builtin_amdgcn_global_load_async_to_lds_b128)
#define HAS_ASYNC 1
#endif

typedef __attribute__((address_space(1))) v4i v4i_glb;
typedef __attribute__((address_space(3))) v4i v4i_lds;

// copy 16 bytes global -> LDS (async DMA when available)
static __device__ inline void async_copy16(half_t* lds_dst, const half_t* gsrc) {
#ifdef HAS_ASYNC
  __builtin_amdgcn_global_load_async_to_lds_b128(
      (v4i_glb*)gsrc, (v4i_lds*)lds_dst, 0, 0);
#else
  *(v4u*)lds_dst = *(const v4u*)gsrc;
#endif
}

static __device__ inline void async_join() {
#ifdef HAS_ASYNC
#if __has_builtin(__builtin_amdgcn_s_wait_asynccnt)
  __builtin_amdgcn_s_wait_asynccnt(0);
#else
  asm volatile("s_wait_asynccnt 0" ::: "memory");
#endif
#endif
  __syncthreads();
}

// A-matrix 16x32 f16 tile load (row-major source, rows = M, ld in halves).
// lane l: row l%16, halves k0+(l/16)*8 .. +7 and k0+16+(l/16)*8 .. +7.
static __device__ inline v16h load_a_tile(const half_t* __restrict__ base, int ld) {
  const int lane = threadIdx.x & 31;
  const half_t* p = base + (lane & 15) * ld + ((lane >> 4) << 3);
  V16U r;
  r.q[0] = *(const v4u*)(p);
  r.q[1] = *(const v4u*)(p + 16);
  return r.h;
}

// B-matrix 32x16 f16 tile load from row-major B^T (rows = N, ld in halves).
// lane l: column l%16 of B, halves k0+(l/16)*8 .. +7 and k0+16+(l/16)*8 .. +7.
static __device__ inline v16h load_b_tile(const half_t* __restrict__ base, int ld) {
  const int lane = threadIdx.x & 31;
  const half_t* p = base + (lane & 15) * ld + ((lane >> 4) << 3);
  V16U r;
  r.q[0] = *(const v4u*)(p);
  r.q[1] = *(const v4u*)(p + 16);
  return r.h;
}

static __device__ inline v8f wmma16(v16h a, v16h b, v8f c) {
  return __builtin_amdgcn_wmma_f32_16x16x32_f16(false, a, false, b, (short)0, c,
                                                false, false);
}

static __device__ inline v8f zero8() {
  v8f z = {0.f, 0.f, 0.f, 0.f, 0.f, 0.f, 0.f, 0.f};
  return z;
}

// ---------------- elementwise / data movement kernels ----------------

__global__ void k_cast(half_t* __restrict__ dst, const float* __restrict__ src, int n) {
  int i = blockIdx.x * blockDim.x + threadIdx.x;
  if (i < n) dst[i] = (half_t)src[i];
}

// cw[w]: [D,D,w] f32 -> packed f16 slice [D,D] for tap j (B^T layout, row=out ch)
__global__ void k_packcw(half_t* __restrict__ dst, const float* __restrict__ src,
                         int w, int j) {
  int i = blockIdx.x * blockDim.x + threadIdx.x;
  if (i < DIM * DIM) {
    int o = i >> 10, ii = i & 1023;
    dst[i] = (half_t)src[(o * DIM + ii) * w + j];
  }
}

// Gather embedding rows for all answer tokens + question tokens, cast to f16.
__global__ void k_gather(half_t* __restrict__ X, const float* __restrict__ emb,
                         const int* __restrict__ a_tok, const int* __restrict__ q_tok) {
  long long e = ((long long)blockIdx.x * blockDim.x + threadIdx.x) * 4;
  const long long total = (long long)(T_TOK + 64) * DIM;
  if (e >= total) return;
  int row = (int)(e >> 10), col = (int)(e & 1023);
  int tok = (row < T_TOK) ? a_tok[row] : q_tok[row - T_TOK];
  const float4 v = *(const float4*)(emb + (size_t)tok * DIM + col);
  v4h o = {(half_t)v.x, (half_t)v.y, (half_t)v.z, (half_t)v.w};
  *(v4h*)(X + (size_t)row * DIM + col) = o;
}

__global__ void k_tq(const half_t* __restrict__ P, half_t* __restrict__ pqT) {
  int i = blockIdx.x * blockDim.x + threadIdx.x;
  if (i < 64 * DIM) {
    int k = i >> 6, q = i & 63;
    pqT[i] = P[(size_t)(T_TOK + q) * DIM + k];
  }
}

__global__ void k_zpad(half_t* __restrict__ sim) {
  int i = blockIdx.x * blockDim.x + threadIdx.x;
  if (i < 16 * DIM) sim[(size_t)T_TOK * DIM + i] = (half_t)0.f;
}

// ---------------- gated projection: P = sigmoid(X W1^T+b1) * tanh(X W2^T+b2) --
// 8 waves / WG; the 16 shared A rows are async-staged into LDS once.

__global__ void __launch_bounds__(256)
k_proj(const half_t* __restrict__ X, const half_t* __restrict__ W1h,
       const half_t* __restrict__ W2h, const float* __restrict__ b1,
       const float* __restrict__ b2, half_t* __restrict__ P) {
  __shared__ __align__(16) half_t sA[16 * DIM];   // 32 KB
  const int r0 = blockIdx.y * 16;
  const int tid = threadIdx.x;
  {
    const half_t* g = X + (size_t)r0 * DIM;
    for (int it = 0; it < 8; ++it) {              // 8 * 256 * 8 halves = 16*1024
      int off = (it * 256 + tid) * 8;
      async_copy16(sA + off, g + off);
    }
  }
  async_join();

  const int wave = tid >> 5;
  const int lane = tid & 31;
  const int n0 = blockIdx.x * 128 + wave * 16;
  v8f c1 = zero8(), c2 = zero8();
  for (int k0 = 0; k0 < DIM; k0 += 32) {
    v16h a   = load_a_tile(sA + k0, DIM);                       // ds_load
    v16h b1t = load_b_tile(W1h + (size_t)n0 * DIM + k0, DIM);
    v16h b2t = load_b_tile(W2h + (size_t)n0 * DIM + k0, DIM);
    c1 = wmma16(a, b1t, c1);
    c2 = wmma16(a, b2t, c2);
  }
  const int col = n0 + (lane & 15);
  const int rbase = r0 + ((lane >> 4) << 3);
  const float bb1 = b1[col], bb2 = b2[col];
  for (int g = 0; g < 8; ++g) {
    float x1 = c1[g] + bb1;
    float x2 = c2[g] + bb2;
    float s = 1.0f / (1.0f + __expf(-x1));
    float t = tanhf(x2);
    P[(size_t)(rbase + g) * DIM + col] = (half_t)(s * t);
  }
}

// ---------------- fused attention: Mt -> softmax -> Y -> sim = pa*Y ----------

__global__ void __launch_bounds__(32)
k_attn(const half_t* __restrict__ P, const half_t* __restrict__ pqT,
       half_t* __restrict__ sim) {
  __shared__ __align__(16) float  smt[16 * 64];
  __shared__ __align__(16) half_t salpha[16 * 64];
  const int r0 = blockIdx.x * 16;
  const int lane = threadIdx.x & 31;

  // Mt = pa[r0:r0+16] @ pq^T  (16 x 64), pq rows at P[T_TOK ..]
  v8f c0 = zero8(), c1 = zero8(), c2 = zero8(), c3 = zero8();
  for (int k0 = 0; k0 < DIM; k0 += 32) {
    v16h a = load_a_tile(P + (size_t)r0 * DIM + k0, DIM);
    c0 = wmma16(a, load_b_tile(P + (size_t)(T_TOK +  0) * DIM + k0, DIM), c0);
    c1 = wmma16(a, load_b_tile(P + (size_t)(T_TOK + 16) * DIM + k0, DIM), c1);
    c2 = wmma16(a, load_b_tile(P + (size_t)(T_TOK + 32) * DIM + k0, DIM), c2);
    c3 = wmma16(a, load_b_tile(P + (size_t)(T_TOK + 48) * DIM + k0, DIM), c3);
  }
  {
    const int col = lane & 15;
    const int mrow = (lane >> 4) << 3;
    for (int g = 0; g < 8; ++g) {
      smt[(mrow + g) * 64 + col +  0] = c0[g];
      smt[(mrow + g) * 64 + col + 16] = c1[g];
      smt[(mrow + g) * 64 + col + 32] = c2[g];
      smt[(mrow + g) * 64 + col + 48] = c3[g];
    }
  }
  __syncthreads();
  // per-token softmax over the 64 question positions
  if (lane < 16) {
    float mx = -1e30f;
    for (int q = 0; q < 64; ++q) mx = fmaxf(mx, smt[lane * 64 + q]);
    float sum = 0.f;
    for (int q = 0; q < 64; ++q) sum += __expf(smt[lane * 64 + q] - mx);
    float inv = 1.0f / sum;
    for (int q = 0; q < 64; ++q)
      salpha[lane * 64 + q] = (half_t)(__expf(smt[lane * 64 + q] - mx) * inv);
  }
  __syncthreads();

  // Y = alpha @ pq (16 x 1024), sim = pa * Y
  const int col16 = lane & 15;
  const int rbase = r0 + ((lane >> 4) << 3);
  for (int t = 0; t < 64; ++t) {
    const int n0 = t * 16;
    v8f c = zero8();
    for (int k0 = 0; k0 < 64; k0 += 32) {
      v16h a = load_a_tile((const half_t*)salpha + k0, 64);
      v16h b = load_b_tile(pqT + (size_t)n0 * 64 + k0, 64);
      c = wmma16(a, b, c);
    }
    const int col = n0 + col16;
    for (int g = 0; g < 8; ++g) {
      float pav = (float)P[(size_t)(rbase + g) * DIM + col];
      sim[(size_t)(rbase + g) * DIM + col] = (half_t)(pav * c[g]);
    }
  }
}

// ---------------- conv (as GEMM) + ReLU + per-answer DMax pooling ------------
// 8 waves / WG cover 128 output channels; the 18 shared sim rows per row-tile
// are async-staged into LDS (tail rows covered by the 16-row zero pad).

__global__ void __launch_bounds__(256)
k_conv(const half_t* __restrict__ sim, const half_t* __restrict__ cwp,
       const float* __restrict__ cb1, const float* __restrict__ cb2,
       const float* __restrict__ cb3, half_t* __restrict__ cat) {
  __shared__ __align__(16) half_t sA[18 * DIM];   // 36 KB
  const int a = blockIdx.y;
  const int widx = blockIdx.z;
  const int w = widx + 1;
  const int sb = (widx == 0) ? 0 : (widx == 1) ? 1 : 3;   // packed slice base
  const float* cb = (widx == 0) ? cb1 : (widx == 1) ? cb2 : cb3;

  const int tid = threadIdx.x;
  const int wave = tid >> 5;
  const int lane = tid & 31;
  const int oc0 = blockIdx.x * 128 + wave * 16;
  const int col = oc0 + (lane & 15);
  const float cbv = cb[col];
  const int mrow = (lane >> 4) << 3;

  v8f runmax = zero8();
  for (int i = 0; i < 32; ++i) {
    const int r0 = a * 512 + i * 16;
    {
      const half_t* g = sim + (size_t)r0 * DIM;
      for (int it = 0; it < 9; ++it) {            // 9 * 256 * 8 halves = 18*1024
        int off = (it * 256 + tid) * 8;
        async_copy16(sA + off, g + off);
      }
    }
    async_join();

    v8f c = zero8();
    for (int j = 0; j < w; ++j) {
      const half_t* Bb = cwp + ((size_t)(sb + j)) * DIM * DIM + (size_t)oc0 * DIM;
      const half_t* Ab = sA + (size_t)j * DIM;
      for (int k0 = 0; k0 < DIM; k0 += 32) {
        v16h av = load_a_tile(Ab + k0, DIM);      // ds_load, shared by 8 waves
        v16h bv = load_b_tile(Bb + k0, DIM);
        c = wmma16(av, bv, c);
      }
    }
    // ragged tail: last tile has only 16-(w-1) valid conv rows per answer
    const int valid = (i == 31) ? (16 - (w - 1)) : 16;
    for (int g = 0; g < 8; ++g) {
      float v = c[g] + cbv;
      v = v > 0.f ? v : 0.f;                      // ReLU (pool floor is 0)
      if (mrow + g < valid) runmax[g] = fmaxf(runmax[g], v);
    }
    __syncthreads();                              // protect sA before re-stage
  }
  float m = runmax[0];
  for (int g = 1; g < 8; ++g) m = fmaxf(m, runmax[g]);
  m = fmaxf(m, __shfl_xor(m, 16, 32));            // combine M=0..7 with M=8..15
  if (lane < 16) cat[(size_t)a * 3072 + widx * DIM + col] = (half_t)m;
}

// ---------------- head: h = tanh(cat @ lin_w^T + lin_b) ---------------------

__global__ void __launch_bounds__(32)
k_head(const half_t* __restrict__ cat, const half_t* __restrict__ linwh,
       const float* __restrict__ lin_b, float* __restrict__ h) {
  const int n0 = blockIdx.x * 16;
  const int m0 = blockIdx.y * 16;
  v8f c = zero8();
  for (int k0 = 0; k0 < 3072; k0 += 32) {
    v16h a = load_a_tile(cat + (size_t)m0 * 3072 + k0, 3072);
    v16h b = load_b_tile(linwh + (size_t)n0 * 3072 + k0, 3072);
    c = wmma16(a, b, c);
  }
  const int lane = threadIdx.x & 31;
  const int col = n0 + (lane & 15);
  const int rbase = m0 + ((lane >> 4) << 3);
  const float bv = lin_b[col];
  for (int g = 0; g < 8; ++g)
    h[(size_t)(rbase + g) * DIM + col] = tanhf(c[g] + bv);
}

__global__ void __launch_bounds__(32)
k_logits(const float* __restrict__ h, const float* __restrict__ soft_w,
         const float* __restrict__ soft_b, float* __restrict__ logits) {
  const int r = blockIdx.x;
  const int lane = threadIdx.x;
  float s = 0.f;
  for (int k = lane; k < DIM; k += 32) s += h[(size_t)r * DIM + k] * soft_w[k];
  for (int off = 16; off >= 1; off >>= 1) s += __shfl_xor(s, off, 32);
  if (lane == 0) logits[r] = s + soft_b[0];
}

__global__ void __launch_bounds__(32)
k_lsm(const float* __restrict__ logits, float* __restrict__ out) {
  const int lane = threadIdx.x;
  float v0 = logits[lane], v1 = logits[lane + 32];
  float m = fmaxf(v0, v1);
  for (int off = 16; off >= 1; off >>= 1) m = fmaxf(m, __shfl_xor(m, off, 32));
  float s = __expf(v0 - m) + __expf(v1 - m);
  for (int off = 16; off >= 1; off >>= 1) s += __shfl_xor(s, off, 32);
  float l = logf(s);
  out[lane] = v0 - m - l;
  out[lane + 32] = v1 - m - l;
}

// ---------------- launcher ---------------------------------------------------

extern "C" void kernel_launch(void* const* d_in, const int* in_sizes, int n_in,
                              void* d_out, int out_size, void* d_ws, size_t ws_size,
                              hipStream_t stream) {
  (void)in_sizes; (void)n_in; (void)out_size; (void)ws_size;
  const int*   data_q  = (const int*)d_in[0];
  const int*   data_as = (const int*)d_in[1];
  const float* emb     = (const float*)d_in[2];
  const float* w1      = (const float*)d_in[3];
  const float* b1      = (const float*)d_in[4];
  const float* w2      = (const float*)d_in[5];
  const float* b2      = (const float*)d_in[6];
  const float* cw1     = (const float*)d_in[7];
  const float* cb1     = (const float*)d_in[8];
  const float* cw2     = (const float*)d_in[9];
  const float* cb2     = (const float*)d_in[10];
  const float* cw3     = (const float*)d_in[11];
  const float* cb3     = (const float*)d_in[12];
  const float* lin_w   = (const float*)d_in[13];
  const float* lin_b   = (const float*)d_in[14];
  const float* soft_w  = (const float*)d_in[15];
  const float* soft_b  = (const float*)d_in[16];

  const int T = T_TOK, R = T + 64;
  char* ws = (char*)d_ws;
  size_t off = 0;
  auto alloc = [&](size_t bytes) -> void* {
    void* p = ws + off;
    off = (off + bytes + 255) & ~(size_t)255;
    return p;
  };
  half_t* X    = (half_t*)alloc((size_t)R * DIM * 2);       // gathered f16 embeds
  half_t* W1h  = (half_t*)alloc((size_t)DIM * DIM * 2);
  half_t* W2h  = (half_t*)alloc((size_t)DIM * DIM * 2);
  half_t* CWP  = (half_t*)alloc((size_t)6 * DIM * DIM * 2); // packed conv taps
  half_t* LINh = (half_t*)alloc((size_t)DIM * 3072 * 2);
  half_t* P    = (half_t*)alloc((size_t)R * DIM * 2);       // pa (+pq tail rows)
  half_t* PQT  = (half_t*)alloc((size_t)DIM * 64 * 2);
  half_t* CAT  = (half_t*)alloc((size_t)64 * 3072 * 2);
  float*  H    = (float*)alloc((size_t)64 * DIM * 4);
  float*  LOG  = (float*)alloc((size_t)256);
  half_t* SIM  = X;  // alias: X dead after k_proj; sim needs (T+16)*DIM <= R*DIM

  // stage 0: casts / packs / gather
  k_gather<<<(R * DIM / 4 + 255) / 256, 256, 0, stream>>>(X, emb, data_as, data_q);
  k_cast<<<(DIM * DIM + 255) / 256, 256, 0, stream>>>(W1h, w1, DIM * DIM);
  k_cast<<<(DIM * DIM + 255) / 256, 256, 0, stream>>>(W2h, w2, DIM * DIM);
  k_cast<<<(DIM * 3072 + 255) / 256, 256, 0, stream>>>(LINh, lin_w, DIM * 3072);
  k_packcw<<<4096, 256, 0, stream>>>(CWP + (size_t)0 * DIM * DIM, cw1, 1, 0);
  k_packcw<<<4096, 256, 0, stream>>>(CWP + (size_t)1 * DIM * DIM, cw2, 2, 0);
  k_packcw<<<4096, 256, 0, stream>>>(CWP + (size_t)2 * DIM * DIM, cw2, 2, 1);
  k_packcw<<<4096, 256, 0, stream>>>(CWP + (size_t)3 * DIM * DIM, cw3, 3, 0);
  k_packcw<<<4096, 256, 0, stream>>>(CWP + (size_t)4 * DIM * DIM, cw3, 3, 1);
  k_packcw<<<4096, 256, 0, stream>>>(CWP + (size_t)5 * DIM * DIM, cw3, 3, 2);

  // stage 1: gated projection for all T+64 rows (8 waves/WG, LDS-staged A)
  k_proj<<<dim3(DIM / 128, R / 16), 256, 0, stream>>>(X, W1h, W2h, b1, b2, P);
  k_tq<<<(64 * DIM + 255) / 256, 256, 0, stream>>>(P, PQT);

  // stage 2: attention + sim (sim aliases X, written after X's last read)
  k_zpad<<<(16 * DIM + 255) / 256, 256, 0, stream>>>(SIM);
  k_attn<<<T / 16, 32, 0, stream>>>(P, PQT, SIM);

  // stage 3: conv-as-GEMM + ReLU + per-answer max pool -> cat [64,3072]
  k_conv<<<dim3(DIM / 128, 64, 3), 256, 0, stream>>>(SIM, CWP, cb1, cb2, cb3, CAT);

  // stage 4: head
  k_head<<<dim3(DIM / 16, 4), 32, 0, stream>>>(CAT, LINh, lin_b, H);
  k_logits<<<64, 32, 0, stream>>>(H, soft_w, soft_b, LOG);
  k_lsm<<<1, 32, 0, stream>>>(LOG, (float*)d_out);
}
